// NARModel_18614388261499
// MI455X (gfx1250) — matrix operand order repaired and checked by reading, not compiled, not run on hardware
//
#include <hip/hip_runtime.h>
#include <hip/hip_bf16.h>
#include <math.h>

typedef __attribute__((ext_vector_type(16))) _Float16 v16h;
typedef __attribute__((ext_vector_type(8)))  _Float16 v8h;
typedef __attribute__((ext_vector_type(8)))  float    v8f;

static constexpr int HD   = 128;      // hidden dim
static constexpr int NB   = 8;        // batch
static constexpr int NN   = 256;      // nodes
static constexpr int ROWS = NB * NN;  // 2048

// ---------------------------------------------------------------------------
// Generic WMMA GEMM:  C[m][n] = act( scale * sum_k A[m][k]*Bt[n][k] + bias[n] )
// A: row-major fp16 (M x K).  Bt: row-major fp16 (Nc x K)  (i.e. B transposed).
// One wave32 computes one 16x16 tile via v_wmma_f32_16x16x32_f16.
// blockIdx.y = batch, with per-batch element strides sA/sB/sC.
// act: 0=none, 1=relu, 2=sigmoid
// ---------------------------------------------------------------------------
__global__ void gemm16(const _Float16* __restrict__ A,
                       const _Float16* __restrict__ Bt,
                       const float* __restrict__ bias,
                       float* __restrict__ C,
                       int M, int Nc, int K,
                       long sA, long sB, long sC,
                       float scale, int act)
{
    const int lane = threadIdx.x & 31;
    const int wave = blockIdx.x * (blockDim.x >> 5) + (threadIdx.x >> 5);
    const int tilesN = Nc >> 4;
    const int tiles  = (M >> 4) * tilesN;
    if (wave >= tiles) return;                 // wave-uniform exit (EXEC stays full)
    const int tm = wave / tilesN;
    const int tn = wave % tilesN;
    const long b = blockIdx.y;
    A  += b * sA;
    Bt += b * sB;
    C  += b * sC;

    const int g = lane >> 4;   // half-wave group
    const int r = lane & 15;

    // A 16x32 operand, this code's contiguous k order: lane row = r; halves e -> K=16g+e (contiguous, matches B)
    const _Float16* arow = A  + (size_t)(tm * 16 + r) * K + 16 * g;
    // B 32x16 operand, same contiguous k order as A: lane col = r; halves e -> K=16g+e (contiguous in Bt row)
    const _Float16* brow = Bt + (size_t)(tn * 16 + r) * K + 16 * g;

    v8f acc = {};
    for (int k0 = 0; k0 < K; k0 += 32) {
        v8h alo = *(const v8h*)(arow + k0);
        v8h ahi = *(const v8h*)(arow + k0 + 8);
        v16h a;
        #pragma unroll
        for (int e = 0; e < 8; ++e) { a[e] = alo[e]; a[e + 8] = ahi[e]; }
        v16h bb = *(const v16h*)(brow + k0);
        acc = __builtin_amdgcn_wmma_f32_16x16x32_f16(
                  false, a, false, bb, (short)0, acc, false, false);
    }

    const int n  = tn * 16 + r;
    const float bv = bias ? bias[n] : 0.0f;
    #pragma unroll
    for (int p = 0; p < 8; ++p) {
        int m   = tm * 16 + p + 8 * g;     // C layout: VGPR p -> M = p + 8*(lane/16)
        float v = acc[p] * scale + bv;
        if (act == 1)      v = fmaxf(v, 0.0f);
        else if (act == 2) v = 1.0f / (1.0f + expf(-v));
        C[(size_t)m * Nc + n] = v;
    }
}

// ---------------------------------------------------------------------------
// Fold m_edge ∘ edge_enc into rank-2 per-edge vectors:
//   u[h] = sqrt(H) * sum_k Me[k][h];  v[h] = sum_k We[k]*Me[k][h];
//   c[h] = sum_k be[k]*Me[k][h] + bm[h]
// ---------------------------------------------------------------------------
__global__ void k_uvc(const float* __restrict__ Me, const float* __restrict__ bm,
                      const float* __restrict__ We, const float* __restrict__ be,
                      float* __restrict__ u, float* __restrict__ v, float* __restrict__ c)
{
    int h = threadIdx.x;
    float su = 0.f, sv = 0.f, sc = 0.f;
    for (int k = 0; k < HD; ++k) {
        float m = Me[k * HD + h];
        su += m; sv += We[k] * m; sc += be[k] * m;
    }
    u[h] = 11.313708498984760f * su;   // sqrt(128)
    v[h] = sv;
    c[h] = sc + bm[h];
}

// W (K x Nc) fp32 -> Wt (Nc x K) fp16
__global__ void k_t16(const float* __restrict__ W, _Float16* __restrict__ Wt,
                      int K, int Nc)
{
    int idx = blockIdx.x * blockDim.x + threadIdx.x;
    if (idx >= K * Nc) return;
    int k = idx / Nc, n = idx % Nc;
    Wt[(size_t)n * K + k] = (_Float16)W[idx];
}

__global__ void k_cvt(const float* __restrict__ x, _Float16* __restrict__ y, int n)
{
    int i = blockIdx.x * blockDim.x + threadIdx.x;
    if (i < n) y[i] = (_Float16)x[i];
}

// Encoder: nodes = pos[n] + nf*Wn0 + bn + pointer_emb[source[b]]
__global__ void k_encoder(const float* __restrict__ nf, const int* __restrict__ src,
                          const float* __restrict__ pos, const float* __restrict__ ptr,
                          const float* __restrict__ wn, const float* __restrict__ bn,
                          float* __restrict__ nodes, _Float16* __restrict__ nodes16)
{
    int idx = blockIdx.x * blockDim.x + threadIdx.x;
    if (idx >= ROWS * HD) return;
    int h = idx & (HD - 1);
    int row = idx >> 7;
    int b = row >> 8;
    int n = row & (NN - 1);
    float v = pos[n * HD + h] + nf[row] * wn[h] + bn[h] + ptr[src[b] * HD + h];
    nodes[idx]   = v;
    nodes16[idx] = (_Float16)v;
}

// agg[b,i,h] = sum_j a_ij * relu(S[b,i,h] + T[b,j,h] + a_ij*u[h] + w_ij*v[h] + c[h])
__global__ void k_agg(const float* __restrict__ S, const float* __restrict__ T,
                      const float* __restrict__ adj, const float* __restrict__ ew,
                      const float* __restrict__ u, const float* __restrict__ v,
                      const float* __restrict__ c, float* __restrict__ agg)
{
    int row = blockIdx.x;            // b*NN + i
    int h   = threadIdx.x;           // 0..127
    int b   = row >> 8;
    __shared__ float sa[NN], sw[NN];
    const float* arow = adj + (size_t)row * NN;
    const float* wrow = ew  + (size_t)row * NN;
    sa[h] = arow[h]; sa[h + 128] = arow[h + 128];
    sw[h] = wrow[h]; sw[h + 128] = wrow[h + 128];
    __syncthreads();
    float s  = S[(size_t)row * HD + h];
    float uh = u[h], vh = v[h], ch = c[h];
    const float* Tb = T + (size_t)b * NN * HD + h;
    float acc = 0.f;
    for (int j = 0; j < NN; ++j) {
        float aj = sa[j], wj = sw[j];
        float x  = s + Tb[(size_t)j * HD] + aj * uh + wj * vh + ch;
        acc += aj * fmaxf(x, 0.f);
    }
    agg[(size_t)row * HD + h] = acc;
}

// cat = [nodes | agg] in fp16
__global__ void k_cat(const float* __restrict__ nodes, const float* __restrict__ agg,
                      _Float16* __restrict__ cat)
{
    int idx = blockIdx.x * blockDim.x + threadIdx.x;
    if (idx >= ROWS * HD) return;
    int row = idx >> 7, h = idx & (HD - 1);
    cat[(size_t)row * 256 + h]       = (_Float16)nodes[idx];
    cat[(size_t)row * 256 + 128 + h] = (_Float16)agg[idx];
}

// nodes = LN( gate*upd + (1-gate)*nodes )   (upd already relu'd, gate already sigmoid'd)
__global__ void k_gateln(const float* __restrict__ upd, const float* __restrict__ gate,
                         const float* __restrict__ lng, const float* __restrict__ lnb,
                         float* __restrict__ nodes, _Float16* __restrict__ nodes16)
{
    int row = blockIdx.x; int h = threadIdx.x;
    size_t i = (size_t)row * HD + h;
    float x = nodes[i];
    float gv = gate[i], uv = upd[i];
    float y = gv * uv + (1.0f - gv) * x;
    __shared__ float red[HD];
    red[h] = y; __syncthreads();
    for (int off = 64; off > 0; off >>= 1) { if (h < off) red[h] += red[h + off]; __syncthreads(); }
    float mu = red[0] * (1.0f / HD);
    __syncthreads();
    float d = y - mu;
    red[h] = d * d; __syncthreads();
    for (int off = 64; off > 0; off >>= 1) { if (h < off) red[h] += red[h + off]; __syncthreads(); }
    float var = red[0] * (1.0f / HD);
    float o = d * rsqrtf(var + 1e-5f) * lng[h] + lnb[h];
    nodes[i]   = o;
    nodes16[i] = (_Float16)o;
}

// reach[row] = h1[row] . w2 + b2
__global__ void k_reach(const float* __restrict__ h1, const float* __restrict__ w2,
                        const float* __restrict__ b2, float* __restrict__ out)
{
    int row = blockIdx.x; int h = threadIdx.x;
    __shared__ float red[HD];
    red[h] = h1[(size_t)row * HD + h] * w2[h];
    __syncthreads();
    for (int off = 64; off > 0; off >>= 1) { if (h < off) red[h] += red[h + off]; __syncthreads(); }
    if (h == 0) out[row] = red[0] + b2[0];
}

// in-place row softmax over 256 columns
__global__ void k_softmax(float* __restrict__ attn)
{
    int row = blockIdx.x; int m = threadIdx.x;
    float* p = attn + (size_t)row * NN;
    float x = p[m];
    __shared__ float red[NN];
    red[m] = x; __syncthreads();
    for (int off = 128; off > 0; off >>= 1) { if (m < off) red[m] = fmaxf(red[m], red[m + off]); __syncthreads(); }
    float mx = red[0]; __syncthreads();
    float e = expf(x - mx);
    red[m] = e; __syncthreads();
    for (int off = 128; off > 0; off >>= 1) { if (m < off) red[m] += red[m + off]; __syncthreads(); }
    p[m] = e / red[0];
}

// ---------------------------------------------------------------------------
extern "C" void kernel_launch(void* const* d_in, const int* in_sizes, int n_in,
                              void* d_out, int out_size, void* d_ws, size_t ws_size,
                              hipStream_t stream)
{
    (void)in_sizes; (void)n_in; (void)out_size; (void)ws_size;
    // ---- inputs (leaves in setup_inputs() insertion order) ----
    const float* node_features = (const float*)d_in[0];
    const float* adjacency     = (const float*)d_in[1];
    const float* edge_weights  = (const float*)d_in[2];
    const int*   source        = (const int*)  d_in[3];
    // d_in[4] = num_steps (== 1)
    const float* position_emb  = (const float*)d_in[5];
    const float* pointer_emb   = (const float*)d_in[6];
    const float* node_enc_w    = (const float*)d_in[7];   // (8,128), row 0 used
    const float* node_enc_b    = (const float*)d_in[8];
    const float* edge_enc_w    = (const float*)d_in[9];   // (1,128)
    const float* edge_enc_b    = (const float*)d_in[10];
    const float *msW[4], *msB[4], *mnW[4], *mnB[4], *meW[4], *meB[4],
                *upW[4], *upB[4], *gaW[4], *gaB[4], *lnG[4], *lnBb[4];
    for (int l = 0; l < 4; ++l) {
        int base = 11 + l * 12;
        msW[l] = (const float*)d_in[base + 0]; msB[l]  = (const float*)d_in[base + 1];
        mnW[l] = (const float*)d_in[base + 2]; mnB[l]  = (const float*)d_in[base + 3];
        meW[l] = (const float*)d_in[base + 4]; meB[l]  = (const float*)d_in[base + 5];
        upW[l] = (const float*)d_in[base + 6]; upB[l]  = (const float*)d_in[base + 7];
        gaW[l] = (const float*)d_in[base + 8]; gaB[l]  = (const float*)d_in[base + 9];
        lnG[l] = (const float*)d_in[base +10]; lnBb[l] = (const float*)d_in[base +11];
    }
    const float* nc1w = (const float*)d_in[59]; const float* nc1b = (const float*)d_in[60];
    const float* nc2w = (const float*)d_in[61]; const float* nc2b = (const float*)d_in[62];
    const float* pqw  = (const float*)d_in[63]; const float* pqb  = (const float*)d_in[64];
    const float* pkw  = (const float*)d_in[65]; const float* pkb  = (const float*)d_in[66];

    // ---- workspace carve ----
    char* ws = (char*)d_ws;
    size_t off = 0;
    auto carve = [&](size_t bytes) -> char* {
        char* p = ws + off; off = (off + bytes + 255) & ~(size_t)255; return p;
    };
    float*    nodes   = (float*)   carve((size_t)ROWS * HD * 4);
    _Float16* nodes16 = (_Float16*)carve((size_t)ROWS * HD * 2);
    float*    Sbuf    = (float*)   carve((size_t)ROWS * HD * 4);
    float*    Tbuf    = (float*)   carve((size_t)ROWS * HD * 4);
    float*    aggb    = (float*)   carve((size_t)ROWS * HD * 4);
    _Float16* cat16   = (_Float16*)carve((size_t)ROWS * 2 * HD * 2);
    float*    updb    = (float*)   carve((size_t)ROWS * HD * 4);
    float*    gateb   = (float*)   carve((size_t)ROWS * HD * 4);
    _Float16* q16     = (_Float16*)carve((size_t)ROWS * HD * 2);
    _Float16* k16     = (_Float16*)carve((size_t)ROWS * HD * 2);
    float*    uvc     = (float*)   carve(4 * 3 * HD * 4);
    _Float16 *mst[4], *mnt[4], *upt[4], *gat[4];
    for (int l = 0; l < 4; ++l) {
        mst[l] = (_Float16*)carve((size_t)HD * HD * 2);
        mnt[l] = (_Float16*)carve((size_t)HD * HD * 2);
        upt[l] = (_Float16*)carve((size_t)2 * HD * HD * 2);
        gat[l] = (_Float16*)carve((size_t)2 * HD * HD * 2);
    }
    _Float16* nc1t = (_Float16*)carve((size_t)HD * HD * 2);
    _Float16* pqt  = (_Float16*)carve((size_t)HD * HD * 2);
    _Float16* pkt  = (_Float16*)carve((size_t)HD * HD * 2);

    // ---- prep: weight transposes (fp16) + edge-fold vectors ----
    const int TB = 256;
    for (int l = 0; l < 4; ++l) {
        k_t16<<<(HD * HD + TB - 1) / TB, TB, 0, stream>>>(msW[l], mst[l], HD, HD);
        k_t16<<<(HD * HD + TB - 1) / TB, TB, 0, stream>>>(mnW[l], mnt[l], HD, HD);
        k_t16<<<(2 * HD * HD + TB - 1) / TB, TB, 0, stream>>>(upW[l], upt[l], 2 * HD, HD);
        k_t16<<<(2 * HD * HD + TB - 1) / TB, TB, 0, stream>>>(gaW[l], gat[l], 2 * HD, HD);
        k_uvc<<<1, HD, 0, stream>>>(meW[l], meB[l], edge_enc_w, edge_enc_b,
                                    uvc + l * 3 * HD, uvc + l * 3 * HD + HD,
                                    uvc + l * 3 * HD + 2 * HD);
    }
    k_t16<<<(HD * HD + TB - 1) / TB, TB, 0, stream>>>(nc1w, nc1t, HD, HD);
    k_t16<<<(HD * HD + TB - 1) / TB, TB, 0, stream>>>(pqw,  pqt,  HD, HD);
    k_t16<<<(HD * HD + TB - 1) / TB, TB, 0, stream>>>(pkw,  pkt,  HD, HD);

    // ---- encoder ----
    k_encoder<<<(ROWS * HD + TB - 1) / TB, TB, 0, stream>>>(
        node_features, source, position_emb, pointer_emb,
        node_enc_w, node_enc_b, nodes, nodes16);

    // ---- processor (num_steps == 1) ----
    const int GEMM_BLOCKS_128 = (ROWS / 16) * (HD / 16) / 4;  // 4 waves/block
    for (int l = 0; l < 4; ++l) {
        gemm16<<<dim3(GEMM_BLOCKS_128, 1), 128, 0, stream>>>(
            nodes16, mst[l], msB[l], Sbuf, ROWS, HD, HD, 0, 0, 0, 1.0f, 0);
        gemm16<<<dim3(GEMM_BLOCKS_128, 1), 128, 0, stream>>>(
            nodes16, mnt[l], mnB[l], Tbuf, ROWS, HD, HD, 0, 0, 0, 1.0f, 0);
        k_agg<<<ROWS, HD, 0, stream>>>(Sbuf, Tbuf, adjacency, edge_weights,
                                       uvc + l * 3 * HD, uvc + l * 3 * HD + HD,
                                       uvc + l * 3 * HD + 2 * HD, aggb);
        k_cat<<<(ROWS * HD + TB - 1) / TB, TB, 0, stream>>>(nodes, aggb, cat16);
        gemm16<<<dim3(GEMM_BLOCKS_128, 1), 128, 0, stream>>>(
            cat16, upt[l], upB[l], updb, ROWS, HD, 2 * HD, 0, 0, 0, 1.0f, 1);
        gemm16<<<dim3(GEMM_BLOCKS_128, 1), 128, 0, stream>>>(
            cat16, gat[l], gaB[l], gateb, ROWS, HD, 2 * HD, 0, 0, 0, 1.0f, 2);
        k_gateln<<<ROWS, HD, 0, stream>>>(updb, gateb, lnG[l], lnBb[l], nodes, nodes16);
    }

    // ---- decoder ----
    float* reach = (float*)d_out;                 // (B*N,)
    float* pred  = (float*)d_out + ROWS;          // (B,N,N)
    gemm16<<<dim3(GEMM_BLOCKS_128, 1), 128, 0, stream>>>(
        nodes16, nc1t, nc1b, Sbuf, ROWS, HD, HD, 0, 0, 0, 1.0f, 1);
    k_reach<<<ROWS, HD, 0, stream>>>(Sbuf, nc2w, nc2b, reach);

    gemm16<<<dim3(GEMM_BLOCKS_128, 1), 128, 0, stream>>>(
        nodes16, pqt, pqb, Tbuf, ROWS, HD, HD, 0, 0, 0, 1.0f, 0);  // q
    gemm16<<<dim3(GEMM_BLOCKS_128, 1), 128, 0, stream>>>(
        nodes16, pkt, pkb, aggb, ROWS, HD, HD, 0, 0, 0, 1.0f, 0);  // k
    k_cvt<<<(ROWS * HD + TB - 1) / TB, TB, 0, stream>>>(Tbuf, q16, ROWS * HD);
    k_cvt<<<(ROWS * HD + TB - 1) / TB, TB, 0, stream>>>(aggb, k16, ROWS * HD);

    // attn[b][n][m] = q_n . k_m / sqrt(H), batched over b
    const int ATTN_BLOCKS = (NN / 16) * (NN / 16) / 4;   // 64
    gemm16<<<dim3(ATTN_BLOCKS, NB), 128, 0, stream>>>(
        q16, k16, nullptr, pred, NN, NN, HD,
        (long)NN * HD, (long)NN * HD, (long)NN * NN,
        0.08838834764831845f /* 1/sqrt(128) */, 0);
    k_softmax<<<ROWS, NN, 0, stream>>>(pred);
}
